// GeneratorBlock_44779329028559
// MI455X (gfx1250) — matrix operand order repaired and checked by reading, not compiled, not run on hardware
//
#include <hip/hip_runtime.h>
#include <hip/hip_bf16.h>

// ---------------- types for WMMA ----------------
typedef __attribute__((ext_vector_type(16))) __bf16 v16bf;
typedef __attribute__((ext_vector_type(8)))  __bf16 v8bf;
typedef __attribute__((ext_vector_type(8)))  float  v8f;

// ---------------- problem constants ----------------
#define BB   8
#define CIN  256
#define FF   128
#define LL   2048
#define LU   4096
#define LUP  4098     // Lu + 2 halo (pad=1 conv folded in)
#define ZZ   256
#define LO2  8192     // final rgb length

// ---------------- workspace layout (bytes, 16B aligned) ----------------
#define N1_OFF   ((size_t)0)                   // [B][F*Lu] f32      16777216
#define N2_OFF   ((size_t)16777216)            //                    16777216
#define XM1_OFF  ((size_t)33554432)            // [B][CIN][LUP] bf16 16785408
#define XM2_OFF  ((size_t)50339840)            // [B][FF][LUP]  bf16  8392704
#define A1_OFF   ((size_t)58732544)            // [FF][768] bf16      196608
#define A2_OFF   ((size_t)58929152)            // [FF][384] bf16       98304
#define XT1_OFF  ((size_t)59027456)            // [B][LU][768] bf16 50331648
#define XT2_OFF  ((size_t)109359104)           // [B][LU][384] bf16 25165824
#define S1_OFF   ((size_t)134524928)           // s1+1 [B][CIN] f32     8192
#define S2_OFF   ((size_t)134533120)           // s2+1 [B][FF]  f32     4096
#define SR_OFF   ((size_t)134537216)           // srgb+1 [B][FF]        4096
#define D1_OFF   ((size_t)134541312)           // demod1 [B][FF]        4096
#define D2_OFF   ((size_t)134545408)           // demod2 [B][FF]        4096
#define RGBP_OFF ((size_t)134549504)           // rgb_pre [B][LU]     131072

// =====================================================================
// zero-fill (halo regions of xm1/xm2 must be zero)
// =====================================================================
__global__ void gb_fill_kernel(unsigned int* p, size_t ndw) {
    size_t i = (size_t)blockIdx.x * blockDim.x + threadIdx.x;
    if (i < ndw) p[i] = 0u;
}

// =====================================================================
// styles: s+1 for all three style branches
// =====================================================================
__global__ void gb_style_kernel(const float* __restrict__ istyle,
                                const float* __restrict__ w1, const float* __restrict__ b1,
                                const float* __restrict__ w2, const float* __restrict__ b2,
                                const float* __restrict__ wr, const float* __restrict__ br,
                                float* __restrict__ s1p, float* __restrict__ s2p,
                                float* __restrict__ srp) {
    int idx = blockIdx.x * 256 + threadIdx.x;        // 0 .. 4095
    const float* w; const float* bias; float* dst; int b, c, stride;
    if (idx < BB * CIN)                    { int t = idx;                    w = w1; bias = b1; dst = s1p; b = t / CIN; c = t % CIN; stride = CIN; }
    else if (idx < BB * CIN + BB * FF)     { int t = idx - BB * CIN;         w = w2; bias = b2; dst = s2p; b = t / FF;  c = t % FF;  stride = FF; }
    else if (idx < BB * CIN + 2 * BB * FF) { int t = idx - BB * CIN - BB*FF; w = wr; bias = br; dst = srp; b = t / FF;  c = t % FF;  stride = FF; }
    else return;
    const float* is = istyle + (size_t)b * ZZ;
    const float* wrow = w + (size_t)c * ZZ;
    float acc = 0.f;
    #pragma unroll 4
    for (int z = 0; z < ZZ; ++z) acc += is[z] * wrow[z];
    dst[(size_t)b * stride + c] = acc + bias[c] + 1.0f;
}

// =====================================================================
// demodulation factors d[b,f] = rsqrt(sum_ck (W*s)^2 + eps)
// =====================================================================
__global__ void gb_demod_kernel(const float* __restrict__ cw1, const float* __restrict__ cw2,
                                const float* __restrict__ s1p, const float* __restrict__ s2p,
                                float* __restrict__ d1, float* __restrict__ d2) {
    int idx = blockIdx.x * 256 + threadIdx.x;        // 0..2047
    if (idx >= 2 * BB * FF) return;
    int which = idx >> 10;
    int t = idx & 1023; int b = t >> 7; int f = t & (FF - 1);
    if (which == 0) {
        const float* wr = cw1 + (size_t)f * CIN * 3;
        const float* sp = s1p + (size_t)b * CIN;
        float sum = 0.f;
        for (int c = 0; c < CIN; ++c) {
            float s = sp[c];
            float a0 = wr[c*3+0]*s, a1 = wr[c*3+1]*s, a2 = wr[c*3+2]*s;
            sum += a0*a0 + a1*a1 + a2*a2;
        }
        d1[b * FF + f] = rsqrtf(sum + 1e-8f);
    } else {
        const float* wr = cw2 + (size_t)f * FF * 3;
        const float* sp = s2p + (size_t)b * FF;
        float sum = 0.f;
        for (int c = 0; c < FF; ++c) {
            float s = sp[c];
            float a0 = wr[c*3+0]*s, a1 = wr[c*3+1]*s, a2 = wr[c*3+2]*s;
            sum += a0*a0 + a1*a1 + a2*a2;
        }
        d2[b * FF + f] = rsqrtf(sum + 1e-8f);
    }
}

// =====================================================================
// repack conv weights f32 -> bf16, layout [F][C*3]
// =====================================================================
__global__ void gb_prepw_kernel(const float* __restrict__ w1, const float* __restrict__ w2,
                                __bf16* __restrict__ A1, __bf16* __restrict__ A2) {
    int idx = blockIdx.x * 256 + threadIdx.x;
    const int N1 = FF * CIN * 3, N2 = FF * FF * 3;
    if (idx < N1) A1[idx] = (__bf16)w1[idx];
    else if (idx < N1 + N2) A2[idx - N1] = (__bf16)w2[idx - N1];
}

// =====================================================================
// upsample 2x (linear, align_corners=False) + style modulate -> bf16 halo buf
// =====================================================================
__global__ void gb_upmod_kernel(const float* __restrict__ x, const float* __restrict__ s1p,
                                __bf16* __restrict__ xm1) {
    size_t idx = (size_t)blockIdx.x * 256 + threadIdx.x;
    size_t total = (size_t)BB * CIN * LU;
    if (idx >= total) return;
    int lo = (int)(idx % LU);
    size_t bc = idx / LU;
    int c = (int)(bc % CIN); int b = (int)(bc / CIN);
    const float* xr = x + bc * LL;
    int j = lo >> 1;
    float v;
    if ((lo & 1) == 0) v = (j == 0) ? xr[0] : 0.25f * xr[j-1] + 0.75f * xr[j];
    else { int j1 = (j + 1 < LL) ? j + 1 : LL - 1; v = 0.75f * xr[j] + 0.25f * xr[j1]; }
    xm1[bc * LUP + 1 + lo] = (__bf16)(v * s1p[(size_t)b * CIN + c]);
}

// =====================================================================
// im2col-transpose: XT[b][n][c*3+k] = xm[b][c][n+k]  (K contiguous per column)
// Reads coalesce across lanes (consecutive n); writes are packed 16B chunks.
// =====================================================================
__global__ __launch_bounds__(256)
void gb_im2colT_kernel(const __bf16* __restrict__ xm, __bf16* __restrict__ XT, int C) {
    int idx = blockIdx.x * 256 + threadIdx.x;        // B*LU columns
    if (idx >= BB * LU) return;
    int n = idx & (LU - 1); int b = idx / LU;
    const __bf16* xb = xm + (size_t)b * C * LUP + n;
    __bf16* dst = XT + (size_t)idx * (size_t)(3 * C);
    for (int c0 = 0; c0 < C; c0 += 8) {
        __bf16 tmp[24];
        #pragma unroll
        for (int cc = 0; cc < 8; ++cc) {
            const __bf16* r = xb + (size_t)(c0 + cc) * LUP;
            tmp[cc*3+0] = r[0]; tmp[cc*3+1] = r[1]; tmp[cc*3+2] = r[2];
        }
        #pragma unroll
        for (int q = 0; q < 3; ++q)
            *(v8bf*)(dst + c0*3 + q*8) = *(const v8bf*)(tmp + q*8);
    }
}

// =====================================================================
// noise projection: out[b, i] = dot(inoise[b,:], W[i,:]) (K=100)
// HBM-bound: W row read once -> 8 batch outputs; inoise staged in LDS.
// =====================================================================
__global__ __launch_bounds__(256)
void gb_noise_kernel(const float* __restrict__ W, const float* __restrict__ inoise,
                     float* __restrict__ out, int rows) {
    __shared__ float lin[BB * 100];
    for (int t = threadIdx.x; t < BB * 100; t += 256) lin[t] = inoise[t];
    __syncthreads();
    int i = blockIdx.x * 256 + threadIdx.x;
    if (i >= rows) return;
    __builtin_prefetch(W + (size_t)i * 100 + 6400, 0, 1);  // 64 rows ahead (quarter block)
    const float4* w4 = (const float4*)(W + (size_t)i * 100);
    float acc[BB];
    #pragma unroll
    for (int b = 0; b < BB; ++b) acc[b] = 0.f;
    #pragma unroll 5
    for (int z4 = 0; z4 < 25; ++z4) {
        float4 w = w4[z4];
        #pragma unroll
        for (int b = 0; b < BB; ++b) {
            float4 v = ((const float4*)lin)[b * 25 + z4];
            acc[b] += w.x*v.x + w.y*v.y + w.z*v.z + w.w*v.w;
        }
    }
    #pragma unroll
    for (int b = 0; b < BB; ++b) out[(size_t)b * rows + i] = acc[b];
}

// =====================================================================
// WMMA implicit-GEMM modulated conv.
//   Y[b,f,l] = sum_K A[f,K] * XT[b,l,K]   (bf16 x bf16 -> f32, K = c*3+k)
// Block = 256 thr = 8 waves; wave w owns F rows [16w,16w+16); block owns 64
// output columns (4 accumulators / wave). A K-slice (128x32 bf16, 8KB) is
// double-buffered in LDS via global_load_async_to_lds_b128 (ASYNCcnt).
// B fragments are single 32B vector loads from XT (K-contiguous).
// epilogue: v = Y*dmod + noise ; leaky_relu(0.2)
//   mode 0: write bf16 xm2 (modulated by next style) for conv2
//   mode 1: write f32 to d_out (final x)
// =====================================================================
__global__ __launch_bounds__(256)
void gb_gemm_kernel(const __bf16* __restrict__ A, const __bf16* __restrict__ XT,
                    const float* __restrict__ noise, const float* __restrict__ dmod,
                    const float* __restrict__ sc_next,
                    __bf16* __restrict__ xm_out, float* __restrict__ f32_out,
                    int Ktot, int mode) {
    __shared__ alignas(16) __bf16 sA[2][FF * 32];   // 2 x 8KB double buffer

    int b    = blockIdx.y;
    int n0   = blockIdx.x * 64;
    int wave = threadIdx.x >> 5;
    int lid  = threadIdx.x & 31;
    int f0   = wave * 16;
    int m    = lid & 15;                    // A row within tile
    int nn   = lid & 15;                    // B/C column within tile
    int abase = (lid < 16) ? 0 : 16;        // A lane-half K base
    int kbase = (lid < 16) ? 0 : 16;        // B lane-half K base

    v8f acc[4];
    #pragma unroll
    for (int nt = 0; nt < 4; ++nt) acc[nt] = (v8f){0.f,0.f,0.f,0.f,0.f,0.f,0.f,0.f};

    // async-copy one 128x32 A slice into LDS buffer `buf`
    auto issueA = [&](int kk, int buf) {
        int ch = threadIdx.x;               // 512 x 16B chunks, 2 per thread
        #pragma unroll
        for (int r = 0; r < 2; ++r, ch += 256) {
            int f = ch >> 2, part = ch & 3;                 // 4 chunks per 64B row
            const __bf16* g = A + (size_t)f * Ktot + kk + part * 8;
            unsigned lds = (unsigned)(size_t)(&sA[buf][0]) + (unsigned)(ch * 16);
            asm volatile("global_load_async_to_lds_b128 %0, %1, off"
                         :: "v"(lds), "v"(g) : "memory");
        }
    };

    issueA(0, 0);
    asm volatile("s_wait_asynccnt 0x0" ::: "memory");
    __syncthreads();

    int ib = 0;
    for (int kk = 0; kk < Ktot; kk += 32, ib ^= 1) {
        if (kk + 32 < Ktot) issueA(kk + 32, ib ^ 1);

        // A fragment from LDS (16 contiguous K per lane half, matching B)
        const __bf16* arow = &sA[ib][(f0 + m) * 32];
        v8bf alo = *(const v8bf*)(arow + abase);
        v8bf ahi = *(const v8bf*)(arow + abase + 8);
        v16bf a;
        #pragma unroll
        for (int i = 0; i < 8; ++i) { a[i] = alo[i]; a[8+i] = ahi[i]; }

        #pragma unroll
        for (int nt = 0; nt < 4; ++nt) {
            // B fragment: 16 contiguous K halves at this column -> one 32B load
            const __bf16* xtc = XT + ((size_t)(b * LU + n0 + nt * 16 + nn)) * Ktot
                                   + kk + kbase;
            v16bf bf = *(const v16bf*)xtc;
            acc[nt] = __builtin_amdgcn_wmma_f32_16x16x32_bf16(
                false, a, false, bf, (short)0, acc[nt], false, false);
        }

        asm volatile("s_wait_asynccnt 0x0" ::: "memory");
        __syncthreads();
    }

    // epilogue (C/D layout: VGPR r -> M = r + 8*(lane>=16), N = lane%16)
    int frow = (lid < 16) ? 0 : 8;
    #pragma unroll
    for (int nt = 0; nt < 4; ++nt) {
        int l = n0 + nt * 16 + nn;
        #pragma unroll
        for (int r = 0; r < 8; ++r) {
            int f = f0 + r + frow;
            float v = acc[nt][r] * dmod[b * FF + f]
                    + noise[((size_t)b * FF + f) * LU + l];
            v = (v > 0.f) ? v : 0.2f * v;
            if (mode == 0)
                xm_out[((size_t)b * FF + f) * LUP + 1 + l] =
                    (__bf16)(v * sc_next[b * FF + f]);
            else
                f32_out[((size_t)b * FF + f) * LU + l] = v;
        }
    }
}

// =====================================================================
// RGB head: 1-tap modulated conv (reduce over F) + residual
// =====================================================================
__global__ void gb_rgbpre_kernel(const float* __restrict__ xfin, const float* __restrict__ prev_rgb,
                                 const float* __restrict__ srp, const float* __restrict__ wrgb,
                                 float* __restrict__ rgb_pre) {
    int idx = blockIdx.x * 256 + threadIdx.x;       // B*LU
    if (idx >= BB * LU) return;
    int l = idx & (LU - 1); int b = idx / LU;
    float sum = prev_rgb[(size_t)b * LU + l];
    const float* xr = xfin + (size_t)b * FF * LU + l;
    #pragma unroll 8
    for (int f = 0; f < FF; ++f)
        sum += wrgb[f] * srp[b * FF + f] * xr[(size_t)f * LU];
    rgb_pre[idx] = sum;
}

__global__ void gb_rgbup_kernel(const float* __restrict__ rgb_pre, float* __restrict__ out) {
    int idx = blockIdx.x * 256 + threadIdx.x;       // B*LO2
    if (idx >= BB * LO2) return;
    int lo = idx & (LO2 - 1); int b = idx / LO2;
    const float* r = rgb_pre + (size_t)b * LU;
    int j = lo >> 1;
    float v;
    if ((lo & 1) == 0) v = (j == 0) ? r[0] : 0.25f * r[j-1] + 0.75f * r[j];
    else { int j1 = (j + 1 < LU) ? j + 1 : LU - 1; v = 0.75f * r[j] + 0.25f * r[j1]; }
    out[idx] = v;
}

// =====================================================================
extern "C" void kernel_launch(void* const* d_in, const int* in_sizes, int n_in,
                              void* d_out, int out_size, void* d_ws, size_t ws_size,
                              hipStream_t stream) {
    const float* x         = (const float*)d_in[0];
    const float* prev_rgb  = (const float*)d_in[1];
    const float* istyle    = (const float*)d_in[2];
    const float* inoise    = (const float*)d_in[3];
    const float* style1_w  = (const float*)d_in[4];
    const float* style1_b  = (const float*)d_in[5];
    const float* conv1_w   = (const float*)d_in[6];
    const float* noise_w1  = (const float*)d_in[7];
    const float* style2_w  = (const float*)d_in[8];
    const float* style2_b  = (const float*)d_in[9];
    const float* conv2_w   = (const float*)d_in[10];
    const float* noise_w2  = (const float*)d_in[11];
    const float* rgb_style_w = (const float*)d_in[12];
    const float* rgb_style_b = (const float*)d_in[13];
    const float* rgb_conv_w  = (const float*)d_in[14];

    char* ws = (char*)d_ws;
    float*  n1   = (float*)(ws + N1_OFF);
    float*  n2   = (float*)(ws + N2_OFF);
    __bf16* xm1  = (__bf16*)(ws + XM1_OFF);
    __bf16* xm2  = (__bf16*)(ws + XM2_OFF);
    __bf16* A1   = (__bf16*)(ws + A1_OFF);
    __bf16* A2   = (__bf16*)(ws + A2_OFF);
    __bf16* XT1  = (__bf16*)(ws + XT1_OFF);
    __bf16* XT2  = (__bf16*)(ws + XT2_OFF);
    float*  s1p  = (float*)(ws + S1_OFF);
    float*  s2p  = (float*)(ws + S2_OFF);
    float*  srp  = (float*)(ws + SR_OFF);
    float*  d1   = (float*)(ws + D1_OFF);
    float*  d2   = (float*)(ws + D2_OFF);
    float*  rgbp = (float*)(ws + RGBP_OFF);

    float* xout   = (float*)d_out;                         // [B][F][LU]
    float* rgbout = xout + (size_t)BB * FF * LU;           // [B][LO2]

    // 1. zero halo buffers (xm1 + xm2 contiguous region)
    {
        size_t ndw = ((size_t)BB * CIN * LUP * 2 + (size_t)BB * FF * LUP * 2) / 4;
        gb_fill_kernel<<<dim3((unsigned)((ndw + 255) / 256)), dim3(256), 0, stream>>>(
            (unsigned int*)xm1, ndw);
    }
    // 2. styles
    gb_style_kernel<<<dim3(16), dim3(256), 0, stream>>>(
        istyle, style1_w, style1_b, style2_w, style2_b, rgb_style_w, rgb_style_b,
        s1p, s2p, srp);
    // 3. demod factors
    gb_demod_kernel<<<dim3(8), dim3(256), 0, stream>>>(conv1_w, conv2_w, s1p, s2p, d1, d2);
    // 4. weight repack to bf16
    gb_prepw_kernel<<<dim3((FF*CIN*3 + FF*FF*3 + 255) / 256), dim3(256), 0, stream>>>(
        conv1_w, conv2_w, A1, A2);
    // 5. upsample + modulate input
    gb_upmod_kernel<<<dim3((unsigned)(((size_t)BB*CIN*LU + 255) / 256)), dim3(256), 0, stream>>>(
        x, s1p, xm1);
    // 6. noise projections (HBM-bound: 2 x 210 MB of weights)
    gb_noise_kernel<<<dim3(FF * LU / 256), dim3(256), 0, stream>>>(noise_w1, inoise, n1, FF * LU);
    gb_noise_kernel<<<dim3(FF * LU / 256), dim3(256), 0, stream>>>(noise_w2, inoise, n2, FF * LU);
    // 7. im2col-transpose for conv1
    gb_im2colT_kernel<<<dim3(BB * LU / 256), dim3(256), 0, stream>>>(xm1, XT1, CIN);
    // 8. conv1 (WMMA, K=768): epilogue writes modulated bf16 for conv2
    gb_gemm_kernel<<<dim3(LU / 64, BB), dim3(256), 0, stream>>>(
        A1, XT1, n1, d1, s2p, xm2, nullptr, CIN * 3, 0);
    // 9. im2col-transpose for conv2
    gb_im2colT_kernel<<<dim3(BB * LU / 256), dim3(256), 0, stream>>>(xm2, XT2, FF);
    // 10. conv2 (WMMA, K=384): epilogue writes final x (f32) to d_out
    gb_gemm_kernel<<<dim3(LU / 64, BB), dim3(256), 0, stream>>>(
        A2, XT2, n2, d2, nullptr, nullptr, xout, FF * 3, 1);
    // 11. rgb head + residual
    gb_rgbpre_kernel<<<dim3(BB * LU / 256), dim3(256), 0, stream>>>(
        xout, prev_rgb, srp, rgb_conv_w, rgbp);
    // 12. final 2x upsample of rgb
    gb_rgbup_kernel<<<dim3(BB * LO2 / 256), dim3(256), 0, stream>>>(rgbp, rgbout);
}
